// BottleneckBlock_37031208026363
// MI455X (gfx1250) — matrix operand order repaired and checked by reading, not compiled, not run on hardware
//
#include <hip/hip_runtime.h>
#include <math.h>

typedef __bf16 v4bf  __attribute__((ext_vector_type(4)));
typedef __bf16 v8bf  __attribute__((ext_vector_type(8)));
typedef __bf16 v16bf __attribute__((ext_vector_type(16)));
typedef float  v4f   __attribute__((ext_vector_type(4)));
typedef float  v8f   __attribute__((ext_vector_type(8)));

#define HW   196
#define WD   14
#define NB   64
#define KC   32
#define TM   64
#define TN   64
#define NTHREADS 128
#define LSTRIDE  40   // bf16 elems per LDS row: 80 bytes -> every row 16B-aligned

union V16 { v16bf v; v8bf h[2]; };

// ---------------------------------------------------------------------------
// Fused GEMM (bf16x3 split of fp32) + bias + BN-stat accumulation.
// Double-buffered LDS; per-chunk register prefetch; b128 staging loads.
//   MODE 0: B[k][m] = x[n][k][m]                       (conv1 1x1, K=1024)
//   MODE 1: B[k][m] = relu(bn1(y1[n][ci][pix+tap]))    (conv2 3x3 implicit, K=2304)
//   MODE 2: B[k][m] = relu(bn2(y2[n][k][m]))           (conv3 1x1, K=256)
// ---------------------------------------------------------------------------
template<int MODE>
__global__ __launch_bounds__(NTHREADS)
void gemm_bf16x3(const float* __restrict__ A,
                 const float* __restrict__ Bsrc,
                 const float* __restrict__ bias,
                 const float* __restrict__ bnA,
                 const float* __restrict__ bnB,
                 float* __restrict__ Y,
                 float* __restrict__ sum,
                 float* __restrict__ sumsq,
                 int M, int K, int Cs)
{
    __shared__ __align__(16) __bf16 Ah[2][TM * LSTRIDE];
    __shared__ __align__(16) __bf16 Al[2][TM * LSTRIDE];
    __shared__ __align__(16) __bf16 Bh[2][TN * LSTRIDE];
    __shared__ __align__(16) __bf16 Bl[2][TN * LSTRIDE];

    const int nBase = blockIdx.x * TN;     // pixel tile
    const int mBase = blockIdx.y * TM;     // output-channel tile
    const int n     = blockIdx.z;          // batch image
    const int t     = threadIdx.x;
    const int lane  = t & 31;
    const int wv    = t >> 5;              // wave id 0..3 -> M sub-strip
    const int lr    = lane & 15;
    const int hs    = lane >> 4;           // half-wave select

    // staging decomposition
    const int arow0 = t >> 3;              // A: rows arow0 + j*16
    const int akq   = t & 7;               // A: float4 slot in 32-wide k
    const int bk4   = t >> 4;              // B: k sub-block (0..7)
    const int bm4   = t & 15;              // B: m sub-block (0..15)

    v8f acc[4];
    #pragma unroll
    for (int s = 0; s < 4; ++s)
        #pragma unroll
        for (int j = 0; j < 8; ++j) acc[s][j] = 0.0f;

    v4f ra[4];          // A prefetch: 4 x float4 (row-contiguous in k)
    v4f rb[4];          // B prefetch: 4x4 block, rb[dk][dm]

    // ---- register prefetch of one K-chunk (all-b128 for modes 0/2) ----
    auto loadAB = [&](int kb) {
        #pragma unroll
        for (int j = 0; j < 4; ++j) {
            int r = arow0 + j * 16;
            ra[j] = *(const v4f*)&A[(size_t)(mBase + r) * K + kb + akq * 4];
        }
        if (MODE == 1) {
            const int pixb = nBase + bm4 * 4;
            #pragma unroll
            for (int dk = 0; dk < 4; ++dk) {
                int kk  = kb + bk4 * 4 + dk;
                int ci  = kk / 9;
                int tap = kk - ci * 9;
                int dh  = tap / 3 - 1;
                int dw  = tap - (tap / 3) * 3 - 1;
                const float* src = &Bsrc[((size_t)n * Cs + ci) * HW];
                float ga = bnA[ci], gb = bnB[ci];
                #pragma unroll
                for (int dm = 0; dm < 4; ++dm) {
                    int pix = pixb + dm;
                    float v = 0.0f;
                    if (pix < HW) {
                        int hh = pix / WD + dh;
                        int ww = pix - (pix / WD) * WD + dw;
                        if (hh >= 0 && hh < WD && ww >= 0 && ww < WD) {
                            float u = src[hh * WD + ww];
                            v = fmaxf(fmaf(ga, u, gb), 0.0f);
                        }
                    }
                    rb[dk][dm] = v;
                }
            }
        } else {
            const int pix = nBase + bm4 * 4;   // float4 never straddles HW (196%4==0)
            #pragma unroll
            for (int dk = 0; dk < 4; ++dk) {
                int kk = kb + bk4 * 4 + dk;
                v4f u = {0.0f, 0.0f, 0.0f, 0.0f};
                if (pix < HW)
                    u = *(const v4f*)&Bsrc[((size_t)n * Cs + kk) * HW + pix];
                if (MODE == 2) {
                    float ga = bnA[kk], gb = bnB[kk];
                    #pragma unroll
                    for (int dm = 0; dm < 4; ++dm)
                        u[dm] = fmaxf(fmaf(ga, u[dm], gb), 0.0f);
                }
                rb[dk] = u;
            }
        }
    };

    // ---- split hi/lo bf16, register-transpose B, b64 LDS stores ----
    auto storeAB = [&](int p) {
        #pragma unroll
        for (int j = 0; j < 4; ++j) {
            int off = (arow0 + j * 16) * LSTRIDE + akq * 4;
            v4bf hi, lo;
            #pragma unroll
            for (int e = 0; e < 4; ++e) {
                float f = ra[j][e];
                __bf16 h = (__bf16)f;
                hi[e] = h;
                lo[e] = (__bf16)(f - (float)h);
            }
            *(v4bf*)&Ah[p][off] = hi;
            *(v4bf*)&Al[p][off] = lo;
        }
        #pragma unroll
        for (int dm = 0; dm < 4; ++dm) {
            int off = (bm4 * 4 + dm) * LSTRIDE + bk4 * 4;
            v4bf hi, lo;
            #pragma unroll
            for (int dk = 0; dk < 4; ++dk) {
                float f = rb[dk][dm];
                __bf16 h = (__bf16)f;
                hi[dk] = h;
                lo[dk] = (__bf16)(f - (float)h);
            }
            *(v4bf*)&Bh[p][off] = hi;
            *(v4bf*)&Bl[p][off] = lo;
        }
    };

    auto compute = [&](int p) {
        // A fragment (ISA 16-bit 16x32 layout): halves at k {0,16} / {8,24}
        const int arow = (wv * 16 + lr) * LSTRIDE + hs * 8;
        V16 ahu, alu;
        ahu.h[0] = *(const v8bf*)&Ah[p][arow];
        ahu.h[1] = *(const v8bf*)&Ah[p][arow + 16];
        alu.h[0] = *(const v8bf*)&Al[p][arow];
        alu.h[1] = *(const v8bf*)&Al[p][arow + 16];
        #pragma unroll
        for (int s = 0; s < 4; ++s) {
            // B fragment: column n=s*16+lr, halves at k {0,16} / {8,24} (matches A)
            const int brow = (s * 16 + lr) * LSTRIDE + hs * 8;
            V16 bhu, blu;
            bhu.h[0] = *(const v8bf*)&Bh[p][brow];
            bhu.h[1] = *(const v8bf*)&Bh[p][brow + 16];
            blu.h[0] = *(const v8bf*)&Bl[p][brow];
            blu.h[1] = *(const v8bf*)&Bl[p][brow + 16];
            // bf16x3: hi*hi + hi*lo + lo*hi (drop lo*lo), f32 accumulate
            acc[s] = __builtin_amdgcn_wmma_f32_16x16x32_bf16(
                false, ahu.v, false, bhu.v, (short)0, acc[s], false, false);
            acc[s] = __builtin_amdgcn_wmma_f32_16x16x32_bf16(
                false, ahu.v, false, blu.v, (short)0, acc[s], false, false);
            acc[s] = __builtin_amdgcn_wmma_f32_16x16x32_bf16(
                false, alu.v, false, bhu.v, (short)0, acc[s], false, false);
        }
    };

    // ---- software pipeline: prefetch regs -> compute -> store other buf ----
    loadAB(0);
    storeAB(0);
    __syncthreads();
    int p = 0;
    for (int kb = KC; kb <= K; kb += KC) {
        const bool more = (kb < K);
        if (more) loadAB(kb);      // global b128 loads issue under WMMA latency
        compute(p);
        if (more) storeAB(p ^ 1);
        __syncthreads();
        p ^= 1;
    }

    // ---- epilogue: bias, store, per-channel BN sum/sumsq ----
    #pragma unroll
    for (int r = 0; r < 8; ++r) {
        int co = mBase + wv * 16 + hs * 8 + r;   // C-layout: VGPR r, half-wave
        float bco = bias[co];
        float vs = 0.0f, vq = 0.0f;
        #pragma unroll
        for (int s = 0; s < 4; ++s) {
            int pix = nBase + s * 16 + lr;
            if (pix < HW) {
                float val = acc[s][r] + bco;
                Y[((size_t)n * M + co) * HW + pix] = val;
                vs += val;
                vq += val * val;
            }
        }
        #pragma unroll
        for (int off = 1; off < 16; off <<= 1) {
            vs += __shfl_xor(vs, off, 32);
            vq += __shfl_xor(vq, off, 32);
        }
        if (lr == 0) {
            atomicAdd(&sum[co], vs);
            atomicAdd(&sumsq[co], vq);
        }
    }
}

// ---------------------------------------------------------------------------
__global__ void zero_kernel(float* __restrict__ p, int nElems)
{
    int i = blockIdx.x * blockDim.x + threadIdx.x;
    if (i < nElems) p[i] = 0.0f;
}

// Fold BN (training-mode batch stats) into per-channel affine a*x+b.
// Optionally multiply by softplus(residual_scale) for the last stage.
__global__ void bn_coeff_kernel(const float* __restrict__ sum,
                                const float* __restrict__ sumsq,
                                const float* __restrict__ gamma,
                                const float* __restrict__ beta,
                                float* __restrict__ a, float* __restrict__ b,
                                int C, float invCnt,
                                const float* __restrict__ rscale)
{
    int c = blockIdx.x * blockDim.x + threadIdx.x;
    if (c >= C) return;
    float mean = sum[c] * invCnt;
    float var  = fmaxf(sumsq[c] * invCnt - mean * mean, 0.0f);
    float s = 1.0f;
    if (rscale) {
        float tv = rscale[0];
        s = (tv > 20.0f) ? tv : log1pf(expf(tv));   // softplus
    }
    float aa = gamma[c] * rsqrtf(var + 1e-5f);
    float bb = beta[c] - aa * mean;
    a[c] = s * aa;
    b[c] = s * bb;
}

// out = relu(x + a3[c]*y3 + b3[c]), fully float4-vectorized (196 % 4 == 0 so
// the channel index is constant within each float4).
__global__ void residual_kernel(const float* __restrict__ x,
                                const float* __restrict__ y3,
                                const float* __restrict__ a,
                                const float* __restrict__ b,
                                float* __restrict__ out)
{
    int i4 = blockIdx.x * blockDim.x + threadIdx.x;
    if (i4 >= (NB * 1024 * HW) / 4) return;
    size_t base = (size_t)i4 * 4;
    int c = (int)(base / HW) & 1023;
    float ga = a[c], gb = b[c];
    v4f xv = *(const v4f*)&x[base];
    v4f yv = *(const v4f*)&y3[base];
    v4f ov;
    #pragma unroll
    for (int e = 0; e < 4; ++e)
        ov[e] = fmaxf(xv[e] + fmaf(ga, yv[e], gb), 0.0f);
    *(v4f*)&out[base] = ov;
}

// ---------------------------------------------------------------------------
extern "C" void kernel_launch(void* const* d_in, const int* in_sizes, int n_in,
                              void* d_out, int out_size, void* d_ws, size_t ws_size,
                              hipStream_t stream)
{
    (void)in_sizes; (void)n_in; (void)out_size; (void)ws_size;
    const float* x   = (const float*)d_in[0];
    const float* w1  = (const float*)d_in[1];
    const float* b1  = (const float*)d_in[2];
    const float* g1  = (const float*)d_in[3];
    const float* be1 = (const float*)d_in[4];
    const float* w2  = (const float*)d_in[5];
    const float* b2  = (const float*)d_in[6];
    const float* g2  = (const float*)d_in[7];
    const float* be2 = (const float*)d_in[8];
    const float* w3  = (const float*)d_in[9];
    const float* b3  = (const float*)d_in[10];
    const float* g3  = (const float*)d_in[11];
    const float* be3 = (const float*)d_in[12];
    const float* rs  = (const float*)d_in[13];
    float* out = (float*)d_out;

    float* ws = (float*)d_ws;
    size_t o = 0;
    float* y1 = ws + o; o += (size_t)NB * 256  * HW;
    float* y2 = ws + o; o += (size_t)NB * 256  * HW;
    float* y3 = ws + o; o += (size_t)NB * 1024 * HW;
    float* sums = ws + o; o += 3072;                   // contiguous stat block
    float* sum1 = sums,        *sq1 = sums + 256;
    float* sum2 = sums + 512,  *sq2 = sums + 768;
    float* sum3 = sums + 1024, *sq3 = sums + 2048;
    float* a1  = ws + o; o += 256;  float* bb1 = ws + o; o += 256;
    float* a2  = ws + o; o += 256;  float* bb2 = ws + o; o += 256;
    float* a3  = ws + o; o += 1024; float* bb3 = ws + o; o += 1024;

    const float invCnt = 1.0f / (float)(NB * HW);      // mean over (N,H,W)
    dim3 blk(NTHREADS);
    dim3 grid12(4, 4, NB);    // 4 pixel tiles x 4 M-tiles (M=256) x 64 batch
    dim3 grid3(4, 16, NB);    // M=1024 -> 16 M-tiles

    zero_kernel<<<(3072 + 255) / 256, 256, 0, stream>>>(sums, 3072);

    // conv1 1x1 reduce: W1[256x1024] x X_n[1024x196]
    gemm_bf16x3<0><<<grid12, blk, 0, stream>>>(w1, x, b1, nullptr, nullptr,
                                               y1, sum1, sq1, 256, 1024, 1024);
    bn_coeff_kernel<<<1, 256, 0, stream>>>(sum1, sq1, g1, be1, a1, bb1,
                                           256, invCnt, nullptr);
    // conv2 3x3 implicit GEMM: K = 256*9, BN1+ReLU fused into B-tile gather
    gemm_bf16x3<1><<<grid12, blk, 0, stream>>>(w2, y1, b2, a1, bb1,
                                               y2, sum2, sq2, 256, 2304, 256);
    bn_coeff_kernel<<<1, 256, 0, stream>>>(sum2, sq2, g2, be2, a2, bb2,
                                           256, invCnt, nullptr);
    // conv3 1x1 expand: W3[1024x256] x relu(bn2(Y2))_n[256x196]
    gemm_bf16x3<2><<<grid3, blk, 0, stream>>>(w3, y2, b3, a2, bb2,
                                              y3, sum3, sq3, 1024, 256, 256);
    // BN3 coeffs with softplus(residual_scale) folded in
    bn_coeff_kernel<<<4, 256, 0, stream>>>(sum3, sq3, g3, be3, a3, bb3,
                                           1024, invCnt, rs);
    // out = relu(x + s*bn3(y3))
    residual_kernel<<<(NB * 1024 * HW / 4 + 255) / 256, 256, 0, stream>>>(
        x, y3, a3, bb3, out);
}
